// SimpleQuantumLLM_9990093931330
// MI455X (gfx1250) — matrix operand order repaired and checked by reading, not compiled, not run on hardware
//
#include <hip/hip_runtime.h>

typedef __attribute__((ext_vector_type(16))) _Float16 v16h;
typedef __attribute__((ext_vector_type(8)))  _Float16 v8h;
typedef __attribute__((ext_vector_type(8)))  float    v8f;

#define TOKENS 4096   // B*L
#define DD     1024
#define LL     2048
#define BBATCH 2
#define VV     32000

// ---------------- block reductions (wave32, 256 threads) ----------------
__device__ __forceinline__ float block_sum(float v, float* s8) {
#pragma unroll
  for (int off = 16; off > 0; off >>= 1) v += __shfl_xor(v, off, 32);
  int wid = threadIdx.x >> 5, lid = threadIdx.x & 31;
  if (lid == 0) s8[wid] = v;
  __syncthreads();
  float r = (threadIdx.x < 8) ? s8[threadIdx.x] : 0.0f;
  if (wid == 0) {
#pragma unroll
    for (int off = 4; off > 0; off >>= 1) r += __shfl_xor(r, off, 32);
    if (lid == 0) s8[0] = r;
  }
  __syncthreads();
  float out = s8[0];
  __syncthreads();
  return out;
}

__device__ __forceinline__ float block_max(float v, float* s8) {
#pragma unroll
  for (int off = 16; off > 0; off >>= 1) v = fmaxf(v, __shfl_xor(v, off, 32));
  int wid = threadIdx.x >> 5, lid = threadIdx.x & 31;
  if (lid == 0) s8[wid] = v;
  __syncthreads();
  float r = (threadIdx.x < 8) ? s8[threadIdx.x] : -3.4e38f;
  if (wid == 0) {
#pragma unroll
    for (int off = 4; off > 0; off >>= 1) r = fmaxf(r, __shfl_xor(r, off, 32));
    if (lid == 0) s8[0] = r;
  }
  __syncthreads();
  float out = s8[0];
  __syncthreads();
  return out;
}

// ---------------- K0: phase table (tiny) ----------------
__global__ __launch_bounds__(256) void phase_kernel(
    const float* __restrict__ pf, float* __restrict__ cph, float* __restrict__ sph) {
  int d = blockIdx.x * 256 + threadIdx.x;
  float ph = tanhf(pf[d]) * 0.01618033988749895f;  // tanh(pf)*PHI*0.01
  cph[d] = cosf(ph);
  sph[d] = sinf(ph);
}

// ---------------- K0b: f32 -> f16 bulk convert (out_W) ----------------
__global__ __launch_bounds__(256) void cvt16_kernel(
    const float* __restrict__ src, _Float16* __restrict__ dst) {
  size_t i = ((size_t)blockIdx.x * 256 + threadIdx.x) * 8;
  const float* s = src + i;
  v8h t;
#pragma unroll
  for (int j = 0; j < 8; ++j) t[j] = (_Float16)s[j];
  *(v8h*)(dst + i) = t;
}

// ---------------- K1: embedding gather + LayerNorm*0.01 -> f16 ----------------
__global__ __launch_bounds__(256) void embed_ln_kernel(
    const int* __restrict__ x, const float* __restrict__ emb, _Float16* __restrict__ xn) {
  __shared__ float s8[8];
  long t = blockIdx.x;
  int idx = x[t];
  const float* e = emb + (long)idx * DD;
  float v[4];
#pragma unroll
  for (int i = 0; i < 4; ++i) v[i] = e[i * 256 + threadIdx.x];
  float s = 0.f, s2 = 0.f;
#pragma unroll
  for (int i = 0; i < 4; ++i) { s += v[i]; s2 += v[i] * v[i]; }
  float mean = block_sum(s, s8) * (1.0f / DD);
  float ex2  = block_sum(s2, s8) * (1.0f / DD);
  float inv  = rsqrtf(ex2 - mean * mean + 1e-5f);
#pragma unroll
  for (int i = 0; i < 4; ++i)
    xn[t * DD + i * 256 + threadIdx.x] = (_Float16)((v[i] - mean) * inv * 0.01f);
}

// ---------------- generic WMMA GEMM: C = A(f16,[M,K]) * B^T (+bias) (+A residual) ----------------
// Block tile 128x128, BK=64, 8 waves as 4(M) x 2(N); wave tile 32x64 = 2x4 fragments.
// B_NK=true : B is [N,K] row-major (weight layout), optionally f32 (converted while staging)
// B_NK=false: B is [K,N] row-major f16 (activation layout), transposed while staging
template <bool BIAS, bool RES, bool B_NK, bool B_F32>
__global__ __launch_bounds__(256) void gemm_wmma_kernel(
    const _Float16* __restrict__ A, const void* __restrict__ Bsrc,
    const float* __restrict__ bias, float* __restrict__ C,
    int M, int N, int K, long sA, long sB, long sC) {
  constexpr int BM = 128, BN = 128, BK = 64;
  constexpr int LDA = BK + 8, LDB = BK + 8;  // 72 f16 = 144 B rows (16B aligned, conflict-free)
  __shared__ __align__(16) _Float16 As[BM * LDA];
  __shared__ __align__(16) _Float16 Bs[BN * LDB];

  const _Float16* Ab = A + (size_t)blockIdx.z * (size_t)sA;
  const char*     Bb = (const char*)Bsrc + (size_t)blockIdx.z * (size_t)sB * (B_F32 ? 4u : 2u);
  float*          Cb = C + (size_t)blockIdx.z * (size_t)sC;

  const int mtile = blockIdx.y * BM, ntile = blockIdx.x * BN;
  const int tid = threadIdx.x, lane = tid & 31, wave = tid >> 5;
  const int waveM = wave & 3, waveN = wave >> 2;  // 4 x 2 wave grid
  const int lmod = lane & 15;
  const int khalfA = (lane < 16) ? 0 : 8;
  const int khalfB = (lane < 16) ? 0 : 8;

  v8f acc[2][4];
#pragma unroll
  for (int fm = 0; fm < 2; ++fm)
#pragma unroll
    for (int fn = 0; fn < 4; ++fn)
#pragma unroll
      for (int r = 0; r < 8; ++r) acc[fm][fn][r] = 0.0f;

  for (int k0 = 0; k0 < K; k0 += BK) {
    // ---- stage A tile (128x64 f16): 1024 v8h chunks, 4 per thread ----
#pragma unroll
    for (int it = 0; it < 4; ++it) {
      int c = it * 256 + tid;
      int row = c >> 3, kc = (c & 7) * 8;
      const _Float16* src = Ab + (size_t)(mtile + row) * K + k0 + kc;
      *(v8h*)(As + row * LDA + kc) = *(const v8h*)src;
      __builtin_prefetch((const void*)(src + BK), 0, 1);  // speculative: dropped if invalid
    }
    // ---- stage B tile (128x64 f16, K-contiguous per N row) ----
    if constexpr (B_NK) {
#pragma unroll
      for (int it = 0; it < 4; ++it) {
        int c = it * 256 + tid;
        int row = c >> 3, kc = (c & 7) * 8;
        if constexpr (B_F32) {
          const float* src = (const float*)Bb + (size_t)(ntile + row) * K + k0 + kc;
          v8h tmp;
#pragma unroll
          for (int i = 0; i < 8; ++i) tmp[i] = (_Float16)src[i];
          *(v8h*)(Bs + row * LDB + kc) = tmp;
          __builtin_prefetch((const void*)(src + BK), 0, 1);
        } else {
          const _Float16* src = (const _Float16*)Bb + (size_t)(ntile + row) * K + k0 + kc;
          *(v8h*)(Bs + row * LDB + kc) = *(const v8h*)src;
          __builtin_prefetch((const void*)(src + BK), 0, 1);
        }
      }
    } else {
      // B is [K,N] f16; vector-load 8 along N, scatter-transpose into Bs[n][k]
      const _Float16* Bh = (const _Float16*)Bb;
#pragma unroll
      for (int it = 0; it < 4; ++it) {
        int c = it * 256 + tid;
        int k = c >> 4, nc = (c & 15) * 8;
        v8h tmp = *(const v8h*)(Bh + (size_t)(k0 + k) * N + ntile + nc);
#pragma unroll
        for (int i = 0; i < 8; ++i) Bs[(nc + i) * LDB + k] = tmp[i];
      }
    }
    __syncthreads();

    // ---- 2 K-steps of 32; per step: 2 A frags, 4 B frags, 8 WMMAs ----
#pragma unroll
    for (int ks = 0; ks < BK; ks += 32) {
      v16h afr[2], bfr[4];
#pragma unroll
      for (int fm = 0; fm < 2; ++fm) {
        const _Float16* ap = As + (waveM * 32 + fm * 16 + lmod) * LDA + ks;
        v8h lo = *(const v8h*)(ap + khalfA);
        v8h hi = *(const v8h*)(ap + 16 + khalfA);
        afr[fm] = __builtin_shufflevector(lo, hi, 0, 1, 2, 3, 4, 5, 6, 7,
                                          8, 9, 10, 11, 12, 13, 14, 15);
      }
#pragma unroll
      for (int fn = 0; fn < 4; ++fn) {
        const _Float16* bp = Bs + (waveN * 64 + fn * 16 + lmod) * LDB + ks + khalfB;
        v8h lo = *(const v8h*)(bp);
        v8h hi = *(const v8h*)(bp + 16);
        bfr[fn] = __builtin_shufflevector(lo, hi, 0, 1, 2, 3, 4, 5, 6, 7,
                                          8, 9, 10, 11, 12, 13, 14, 15);
      }
#pragma unroll
      for (int fm = 0; fm < 2; ++fm)
#pragma unroll
        for (int fn = 0; fn < 4; ++fn)
          acc[fm][fn] = __builtin_amdgcn_wmma_f32_16x16x32_f16(
              false, afr[fm], false, bfr[fn], (short)0, acc[fm][fn], false, false);
    }
    __syncthreads();
  }

  // ---- epilogue ----
  const int rowAdd = (lane >= 16) ? 8 : 0;
#pragma unroll
  for (int fm = 0; fm < 2; ++fm)
#pragma unroll
    for (int fn = 0; fn < 4; ++fn)
#pragma unroll
      for (int r = 0; r < 8; ++r) {
        int row = mtile + waveM * 32 + fm * 16 + rowAdd + r;
        int col = ntile + waveN * 64 + fn * 16 + lmod;
        float v = acc[fm][fn][r];
        if constexpr (BIAS) v += bias[col];
        if constexpr (RES)  v += (float)Ab[(size_t)row * K + col];
        Cb[(size_t)row * N + col] = v;
      }
}

// ---------------- K3: quantum-state elementwise + double LN + row sums ----------------
__global__ __launch_bounds__(256) void state_kernel(
    const float* __restrict__ gr, const float* __restrict__ ex,
    const float* __restrict__ cph, const float* __restrict__ sph,
    _Float16* __restrict__ qr, _Float16* __restrict__ qi,
    float* __restrict__ qsr, float* __restrict__ qsi) {
  __shared__ float s8[8];
  long t = blockIdx.x;
  const float* g = gr + t * DD;
  const float* e = ex + t * DD;
  float cr[4], ci[4];
  float s1 = 0, s2 = 0, s3 = 0, s4 = 0;
#pragma unroll
  for (int i = 0; i < 4; ++i) {
    int d = i * 256 + threadIdx.x;
    float G = g[d], E = e[d];
    float r_ = G + E * cph[d];
    float i_ = E * sph[d];
    float nr = sqrtf(r_ * r_ + i_ * i_ + 1e-8f);
    r_ /= nr; i_ /= nr;
    cr[i] = r_; ci[i] = i_;
    s1 += r_; s2 += r_ * r_; s3 += i_; s4 += i_ * i_;
  }
  float mr = block_sum(s1, s8) * (1.0f / DD);
  float vr = block_sum(s2, s8) * (1.0f / DD) - mr * mr;
  float mi = block_sum(s3, s8) * (1.0f / DD);
  float vi = block_sum(s4, s8) * (1.0f / DD) - mi * mi;
  float ir = rsqrtf(vr + 1e-5f), ii = rsqrtf(vi + 1e-5f);
  s1 = s2 = s3 = s4 = 0;
#pragma unroll
  for (int i = 0; i < 4; ++i) {
    cr[i] = (cr[i] - mr) * ir;
    ci[i] = (ci[i] - mi) * ii;
    s1 += cr[i]; s2 += cr[i] * cr[i]; s3 += ci[i]; s4 += ci[i] * ci[i];
  }
  // second LayerNorm (q_norm), *0.01
  mr = block_sum(s1, s8) * (1.0f / DD);
  vr = block_sum(s2, s8) * (1.0f / DD) - mr * mr;
  mi = block_sum(s3, s8) * (1.0f / DD);
  vi = block_sum(s4, s8) * (1.0f / DD) - mi * mi;
  ir = rsqrtf(vr + 1e-5f); ii = rsqrtf(vi + 1e-5f);
  float sqr = 0, sqi = 0;
#pragma unroll
  for (int i = 0; i < 4; ++i) {
    int d = i * 256 + threadIdx.x;
    float yr = (cr[i] - mr) * ir * 0.01f;
    float yi = (ci[i] - mi) * ii * 0.01f;
    qr[t * DD + d] = (_Float16)yr;
    qi[t * DD + d] = (_Float16)yi;
    sqr += yr; sqi += yi;
  }
  sqr = block_sum(sqr, s8);
  sqi = block_sum(sqi, s8);
  if (threadIdx.x == 0) {
    qsr[t] = sqr * 0.03125f;  // * D^-0.5 = 1/32
    qsi[t] = sqi * 0.03125f;
  }
}

// ---------------- K4: attention row softmax, cos(atan2(y,x)) = x/hypot ----------------
__global__ __launch_bounds__(256) void attn_kernel(
    const float* __restrict__ qsr, const float* __restrict__ qsi, _Float16* __restrict__ attn) {
  __shared__ float s8[8];
  int q = blockIdx.x;           // 0..4095 (includes batch)
  int b = q >> 11;
  const float* rb = qsr + (long)b * LL;
  const float* ib = qsi + (long)b * LL;
  float sr = qsr[q], si = qsi[q];
  float sc[8];
  float mx = -3.4e38f;
#pragma unroll
  for (int i = 0; i < 8; ++i) {
    int k = i * 256 + threadIdx.x;
    float xr = sr - rb[k] + 1e-8f;
    float xi = si - ib[k] + 1e-8f;
    float s = 10.0f * xr * rsqrtf(xr * xr + xi * xi);
    sc[i] = s;
    mx = fmaxf(mx, s);
  }
  mx = block_max(mx, s8);
  float sum = 0.f;
#pragma unroll
  for (int i = 0; i < 8; ++i) { sc[i] = __expf(sc[i] - mx); sum += sc[i]; }
  sum = block_sum(sum, s8);
  float inv = 1.0f / sum;
#pragma unroll
  for (int i = 0; i < 8; ++i)
    attn[(long)q * LL + i * 256 + threadIdx.x] = (_Float16)(sc[i] * inv);
}

// ---------------- K6: LN(0.1*attn_out) -> combined f16 [t, 2D] ----------------
__global__ __launch_bounds__(256) void postln_kernel(
    const float* __restrict__ orr, const float* __restrict__ oir, _Float16* __restrict__ cmb) {
  __shared__ float s8[8];
  long t = blockIdx.x;
  float vr[4], vi[4];
  float s1 = 0, s2 = 0, s3 = 0, s4 = 0;
#pragma unroll
  for (int i = 0; i < 4; ++i) {
    int d = i * 256 + threadIdx.x;
    vr[i] = orr[t * DD + d] * 0.1f;
    vi[i] = oir[t * DD + d] * 0.1f;
    s1 += vr[i]; s2 += vr[i] * vr[i]; s3 += vi[i]; s4 += vi[i] * vi[i];
  }
  float mr = block_sum(s1, s8) * (1.0f / DD);
  float Vr = block_sum(s2, s8) * (1.0f / DD) - mr * mr;
  float mi = block_sum(s3, s8) * (1.0f / DD);
  float Vi = block_sum(s4, s8) * (1.0f / DD) - mi * mi;
  float ir = rsqrtf(Vr + 1e-5f), ii = rsqrtf(Vi + 1e-5f);
#pragma unroll
  for (int i = 0; i < 4; ++i) {
    int d = i * 256 + threadIdx.x;
    cmb[t * (2 * DD) + d]      = (_Float16)((vr[i] - mr) * ir);
    cmb[t * (2 * DD) + DD + d] = (_Float16)((vi[i] - mi) * ii);
  }
}

// ---------------- launcher ----------------
extern "C" void kernel_launch(void* const* d_in, const int* in_sizes, int n_in,
                              void* d_out, int out_size, void* d_ws, size_t ws_size,
                              hipStream_t stream) {
  (void)in_sizes; (void)n_in; (void)out_size;
  const int*   x   = (const int*)d_in[0];
  const float* emb = (const float*)d_in[1];
  const float* gW  = (const float*)d_in[2];
  const float* gb  = (const float*)d_in[3];
  const float* eW  = (const float*)d_in[4];
  const float* eb  = (const float*)d_in[5];
  const float* pf  = (const float*)d_in[6];
  const float* oW  = (const float*)d_in[7];
  const float* ob  = (const float*)d_in[8];
  float* out = (float*)d_out;

  char* w = (char*)d_ws;
  _Float16* xn16  = (_Float16*)(w);                         // 8 MB
  _Float16* qr16  = (_Float16*)(w + (8ull  << 20));         // 8 MB
  _Float16* qi16  = (_Float16*)(w + (16ull << 20));         // 8 MB
  _Float16* cmb16 = (_Float16*)(w + (24ull << 20));         // 16 MB
  _Float16* att16 = (_Float16*)(w + (40ull << 20));         // 16 MB
  float*    grf   = (float*)(w + (56ull << 20));            // 16 MB (ground / out_r)
  float*    exf   = (float*)(w + (72ull << 20));            // 16 MB (excited / out_i)
  float*    qsr   = (float*)(w + (88ull << 20));            // 16 KB
  float*    qsi   = (float*)(w + (88ull << 20) + (64u << 10));
  float*    cph   = (float*)(w + (88ull << 20) + (128u << 10));  // 4 KB
  float*    sph   = (float*)(w + (88ull << 20) + (192u << 10));  // 4 KB
  _Float16* w16   = (_Float16*)(w + (89ull << 20));         // 131 MB (optional)
  const size_t need_w16 = (89ull << 20) + (size_t)VV * 2 * DD * sizeof(_Float16);
  const bool   use_w16  = (ws_size >= need_w16);

  // 0. phase table + (optional) out_W f32->f16 pre-conversion (makes W L2-resident)
  phase_kernel<<<DD / 256, 256, 0, stream>>>(pf, cph, sph);
  if (use_w16)
    cvt16_kernel<<<(VV * 2 * DD) / (256 * 8), 256, 0, stream>>>(oW, w16);

  // 1. gather + LN -> xn (f16)
  embed_ln_kernel<<<TOKENS, 256, 0, stream>>>(x, emb, xn16);

  // 2. ground/excited = xn @ W^T + b + xn   (WMMA, weight f32 converted in-stage)
  dim3 g1(DD / 128, TOKENS / 128, 1);
  gemm_wmma_kernel<true, true, true, true><<<g1, 256, 0, stream>>>(
      xn16, gW, gb, grf, TOKENS, DD, DD, 0, 0, 0);
  gemm_wmma_kernel<true, true, true, true><<<g1, 256, 0, stream>>>(
      xn16, eW, eb, exf, TOKENS, DD, DD, 0, 0, 0);

  // 3. state mix + double LN + row sums
  state_kernel<<<TOKENS, 256, 0, stream>>>(grf, exf, cph, sph, qr16, qi16, qsr, qsi);

  // 4. attention softmax rows
  attn_kernel<<<TOKENS, 256, 0, stream>>>(qsr, qsi, att16);

  // 5. out = attn @ q  (batched over grid.z, B operand is [K,N] f16)
  dim3 g2(DD / 128, LL / 128, BBATCH);
  gemm_wmma_kernel<false, false, false, false><<<g2, 256, 0, stream>>>(
      att16, qr16, nullptr, grf, LL, DD, LL,
      (long)LL * LL, (long)LL * DD, (long)LL * DD);
  gemm_wmma_kernel<false, false, false, false><<<g2, 256, 0, stream>>>(
      att16, qi16, nullptr, exf, LL, DD, LL,
      (long)LL * LL, (long)LL * DD, (long)LL * DD);

  // 6. LN(0.1*out) -> combined f16
  postln_kernel<<<TOKENS, 256, 0, stream>>>(grf, exf, cmb16);

  // 7. logits = combined @ out_W^T + out_b   (the 537-GFLOP GEMM)
  dim3 g3(VV / 128, TOKENS / 128, 1);
  if (use_w16)
    gemm_wmma_kernel<true, false, true, false><<<g3, 256, 0, stream>>>(
        cmb16, w16, ob, out, TOKENS, VV, 2 * DD, 0, 0, 0);
  else
    gemm_wmma_kernel<true, false, true, true><<<g3, 256, 0, stream>>>(
        cmb16, oW, ob, out, TOKENS, VV, 2 * DD, 0, 0, 0);
}
